// GAT_82454782148697
// MI455X (gfx1250) — matrix operand order repaired and checked by reading, not compiled, not run on hardware
//
#include <hip/hip_runtime.h>
#include <hip/hip_bf16.h>
#include <cmath>

typedef __attribute__((ext_vector_type(16))) _Float16 v16h;
typedef __attribute__((ext_vector_type(8)))  _Float16 v8h;
typedef __attribute__((ext_vector_type(8)))  float    v8f;

#define NEG_SLOPE 0.2f
#define BN_EPS    1e-5f

static inline int cdiv(long long a, long long b) { return (int)((a + b - 1) / b); }

// ---------- order-preserving float<->int encoding for atomicMax ----------
__device__ __forceinline__ int enc_f(float f) {
    int i = __float_as_int(f);
    return (i >= 0) ? i : (i ^ 0x7FFFFFFF);
}
__device__ __forceinline__ float dec_f(int k) {
    int i = (k >= 0) ? k : (k ^ 0x7FFFFFFF);
    return __int_as_float(i);
}

// ---------- conversions ----------
__global__ void k_cvt_f16(const float* __restrict__ src, _Float16* __restrict__ dst, int n) {
    int i = blockIdx.x * blockDim.x + threadIdx.x;
    if (i < n) dst[i] = (_Float16)src[i];
}

// W [K=128, Ncol=128] row-major fp32 -> Wt f16 [Ncol, K] (transposed)
__global__ void k_cvt_wt(const float* __restrict__ W, _Float16* __restrict__ Wt) {
    int i = blockIdx.x * blockDim.x + threadIdx.x;
    if (i < 128 * 128) {
        int k = i >> 7, n = i & 127;
        Wt[n * 128 + k] = (_Float16)W[i];
    }
}

// ---------- fused WMMA GEMM + attention-dot epilogue ----------
// Xh f16 [N,128] row-major. Wt f16 [128 outcols, 128 k] (transposed weight).
// Each wave: 16-node x 16-col tile; col tile == head (C=16).
// Epilogue: Hlin tile store + per-row dot with att_src/att_dst via shfl reduction.
__global__ __launch_bounds__(256) void k_gemm_att(
    const _Float16* __restrict__ Xh, const _Float16* __restrict__ Wt,
    const float* __restrict__ attS, const float* __restrict__ attD,
    float* __restrict__ Hlin, float* __restrict__ a_src, float* __restrict__ a_dst, int N)
{
    const int lane = threadIdx.x & 31;
    const int head = threadIdx.x >> 5;          // 8 waves -> 8 heads (col tiles)
    const int m0   = blockIdx.x * 16;           // node tile base
    const int mloc = lane & 15;
    const int p    = lane >> 4;                 // half-wave select
    int rowL = m0 + mloc; if (rowL >= N) rowL = N - 1;   // clamped load row
    const int nBase = head * 16;
    const int ncol  = lane & 15;

    v8f c = {};
    #pragma unroll
    for (int kb = 0; kb < 128; kb += 32) {
        // A fragment (16x32 f16): lane m=l&15; halves[0..7]=K(kb+8p+j), halves[8..15]=K(kb+16+8p+j)
        const v8h alo = *(const v8h*)(Xh + (size_t)rowL * 128 + kb + 8 * p);
        const v8h ahi = *(const v8h*)(Xh + (size_t)rowL * 128 + kb + 16 + 8 * p);
        v16h a;
        #pragma unroll
        for (int j = 0; j < 8; ++j) { a[j] = alo[j]; a[8 + j] = ahi[j]; }
        // B fragment (32x16 f16): lane n=l&15; halves[0..7]=K(kb+8p+j), halves[8..15]=K(kb+16+8p+j)
        const v8h blo = *(const v8h*)(Wt + (size_t)(nBase + ncol) * 128 + kb + 8 * p);
        const v8h bhi = *(const v8h*)(Wt + (size_t)(nBase + ncol) * 128 + kb + 16 + 8 * p);
        v16h b;
        #pragma unroll
        for (int j = 0; j < 8; ++j) { b[j] = blo[j]; b[8 + j] = bhi[j]; }
        c = __builtin_amdgcn_wmma_f32_16x16x32_f16(false, a, false, b, (short)0, c, false, false);
    }

    // D layout: VGPR r holds row (r + 8p), col = ncol.
    // Wave-uniform bounds test: full tiles take the straight-line store path.
    const bool full = (m0 + 16 <= N);
    if (full) {
        #pragma unroll
        for (int r = 0; r < 8; ++r)
            Hlin[(size_t)(m0 + r + 8 * p) * 128 + nBase + ncol] = c[r];
    } else {
        #pragma unroll
        for (int r = 0; r < 8; ++r) {
            int rr = m0 + r + 8 * p;
            if (rr < N) Hlin[(size_t)rr * 128 + nBase + ncol] = c[r];
        }
    }

    // attention partial dots, reduced across the 16 lanes of each half-wave
    const float as = attS[head * 16 + ncol];
    const float ad = attD[head * 16 + ncol];
    float ps[8], pd[8];
    #pragma unroll
    for (int r = 0; r < 8; ++r) { ps[r] = c[r] * as; pd[r] = c[r] * ad; }
    #pragma unroll
    for (int mask = 8; mask >= 1; mask >>= 1) {
        #pragma unroll
        for (int r = 0; r < 8; ++r) {
            ps[r] += __shfl_xor(ps[r], mask, 16);
            pd[r] += __shfl_xor(pd[r], mask, 16);
        }
    }
    if (ncol == 0) {
        if (full) {
            #pragma unroll
            for (int r = 0; r < 8; ++r) {
                int rr = m0 + r + 8 * p;
                a_src[(size_t)rr * 8 + head] = ps[r];
                a_dst[(size_t)rr * 8 + head] = pd[r];
            }
        } else {
            #pragma unroll
            for (int r = 0; r < 8; ++r) {
                int rr = m0 + r + 8 * p;
                if (rr < N) {
                    a_src[(size_t)rr * 8 + head] = ps[r];
                    a_dst[(size_t)rr * 8 + head] = pd[r];
                }
            }
        }
    }
}

// ---------- per-layer scratch init ----------
__global__ void k_init(int* __restrict__ segmax, float* __restrict__ segsum,
                       float* __restrict__ accum, int n8, int n128) {
    int i = blockIdx.x * blockDim.x + threadIdx.x;
    if (i < n8) { segmax[i] = enc_f(-INFINITY); segsum[i] = 0.f; }
    if (i < n128) accum[i] = 0.f;
}

__global__ void k_zero2(float* __restrict__ a, float* __restrict__ b, int n) {
    int i = blockIdx.x * blockDim.x + threadIdx.x;
    if (i < n) { a[i] = 0.f; b[i] = 0.f; }
}

// ---------- edge passes (thread = (edge, head)); edges >= E are self-loops ----------
__device__ __forceinline__ void edge_sd(const int* __restrict__ srcI, const int* __restrict__ dstI,
                                        long long e, int E, int& s, int& d) {
    if (e < E) { s = srcI[e]; d = dstI[e]; }
    else       { s = d = (int)(e - E); }
}

__global__ void k_edge_max(const int* __restrict__ srcI, const int* __restrict__ dstI, int E, int N,
                           const float* __restrict__ a_src, const float* __restrict__ a_dst,
                           int* __restrict__ segmax) {
    long long t = (long long)blockIdx.x * blockDim.x + threadIdx.x;
    if (t >= (long long)(E + N) * 8) return;
    int h = (int)(t & 7); long long e = t >> 3;
    int s, d; edge_sd(srcI, dstI, e, E, s, d);
    float v = a_src[(size_t)s * 8 + h] + a_dst[(size_t)d * 8 + h];
    v = v > 0.f ? v : NEG_SLOPE * v;
    atomicMax(&segmax[(size_t)d * 8 + h], enc_f(v));
}

__global__ void k_edge_sum(const int* __restrict__ srcI, const int* __restrict__ dstI, int E, int N,
                           const float* __restrict__ a_src, const float* __restrict__ a_dst,
                           const int* __restrict__ segmax, float* __restrict__ segsum) {
    long long t = (long long)blockIdx.x * blockDim.x + threadIdx.x;
    if (t >= (long long)(E + N) * 8) return;
    int h = (int)(t & 7); long long e = t >> 3;
    int s, d; edge_sd(srcI, dstI, e, E, s, d);
    float v = a_src[(size_t)s * 8 + h] + a_dst[(size_t)d * 8 + h];
    v = v > 0.f ? v : NEG_SLOPE * v;
    atomicAdd(&segsum[(size_t)d * 8 + h], __expf(v - dec_f(segmax[(size_t)d * 8 + h])));
}

__global__ void k_edge_aggr(const int* __restrict__ srcI, const int* __restrict__ dstI, int E, int N,
                            const float* __restrict__ a_src, const float* __restrict__ a_dst,
                            const int* __restrict__ segmax, const float* __restrict__ segsum,
                            const float* __restrict__ Hlin, float* __restrict__ accum) {
    long long t = (long long)blockIdx.x * blockDim.x + threadIdx.x;
    if (t >= (long long)(E + N) * 8) return;
    int h = (int)(t & 7); long long e = t >> 3;
    int s, d; edge_sd(srcI, dstI, e, E, s, d);
    const float* hrow = Hlin + (size_t)s * 128 + h * 16;
    __builtin_prefetch(hrow, 0, 0);                       // global_prefetch_b8
    float v = a_src[(size_t)s * 8 + h] + a_dst[(size_t)d * 8 + h];
    v = v > 0.f ? v : NEG_SLOPE * v;
    float alpha = __expf(v - dec_f(segmax[(size_t)d * 8 + h])) /
                  (segsum[(size_t)d * 8 + h] + 1e-16f);
    float* orow = accum + (size_t)d * 128 + h * 16;
    #pragma unroll
    for (int cc = 0; cc < 16; ++cc) atomicAdd(&orow[cc], hrow[cc] * alpha);
}

// ---------- batch-norm column statistics (block-local LDS reduce + few atomics) ----------
template <int F>
__global__ __launch_bounds__(256) void k_colstats(const float* __restrict__ X, int Nrows,
                                                  float* __restrict__ colsum, float* __restrict__ colsq) {
    __shared__ float ssum[256], ssq[256];
    const int tid = threadIdx.x;
    const int col = tid % F;
    const int grp = tid / F;
    const int G   = 256 / F;
    float s = 0.f, q = 0.f;
    for (long long r = (long long)blockIdx.x * G + grp; r < Nrows; r += (long long)gridDim.x * G) {
        float v = X[r * F + col];
        s += v; q += v * v;
    }
    ssum[tid] = s; ssq[tid] = q;
    __syncthreads();
    for (int g = G >> 1; g >= 1; g >>= 1) {
        if (grp < g) { ssum[tid] += ssum[tid + g * F]; ssq[tid] += ssq[tid + g * F]; }
        __syncthreads();
    }
    if (tid < F) { atomicAdd(&colsum[tid], ssum[tid]); atomicAdd(&colsq[tid], ssq[tid]); }
}

// ---------- BN + ELU (layer 0 output) ----------
__global__ void k_bn_elu(const float* __restrict__ X, const float* __restrict__ colsum,
                         const float* __restrict__ colsq, const float* __restrict__ gamma,
                         const float* __restrict__ beta, float* __restrict__ Y, int Nrows) {
    long long i = (long long)blockIdx.x * blockDim.x + threadIdx.x;
    if (i >= (long long)Nrows * 128) return;
    int c = (int)(i & 127);
    float inv = 1.f / (float)Nrows;
    float mu  = colsum[c] * inv;
    float var = colsq[c] * inv - mu * mu;
    float v = (X[i] - mu) * rsqrtf(var + BN_EPS) * gamma[c] + beta[c];
    Y[i] = v > 0.f ? v : (__expf(v) - 1.f);
}

// ---------- head mean (layer 1, concat=False) ----------
__global__ void k_head_mean(const float* __restrict__ accum, float* __restrict__ hm, int N) {
    int i = blockIdx.x * blockDim.x + threadIdx.x;
    if (i >= N * 16) return;
    int n = i >> 4, c = i & 15;
    float s = 0.f;
    #pragma unroll
    for (int h = 0; h < 8; ++h) s += accum[(size_t)n * 128 + h * 16 + c];
    hm[i] = 0.125f * s;
}

// ---------- final: BN (no ELU) + 16x2 classifier ----------
__global__ void k_final(const float* __restrict__ hm, const float* __restrict__ colsum,
                        const float* __restrict__ colsq, const float* __restrict__ gamma,
                        const float* __restrict__ beta, const float* __restrict__ Wc,
                        const float* __restrict__ bc, float* __restrict__ out, int Nrows) {
    int n = blockIdx.x * blockDim.x + threadIdx.x;
    if (n >= Nrows) return;
    float inv = 1.f / (float)Nrows;
    float l0 = bc[0], l1 = bc[1];
    #pragma unroll
    for (int c = 0; c < 16; ++c) {
        float mu  = colsum[c] * inv;
        float var = colsq[c] * inv - mu * mu;
        float v = (hm[(size_t)n * 16 + c] - mu) * rsqrtf(var + BN_EPS) * gamma[c] + beta[c];
        l0 += v * Wc[c * 2 + 0];
        l1 += v * Wc[c * 2 + 1];
    }
    out[(size_t)n * 2 + 0] = l0;
    out[(size_t)n * 2 + 1] = l1;
}

// ---------- one full GAT layer (enqueue only; all on stream) ----------
static void run_layer(const float* Xf32, const float* W, const float* attS, const float* attD,
                      _Float16* xh, _Float16* wth, float* Hlin, float* a_src, float* a_dst,
                      int* segmax, float* segsum, float* accum,
                      const int* srcI, const int* dstI, int N, int E, hipStream_t stream) {
    const int nx = N * 128;
    k_cvt_f16<<<cdiv(nx, 256), 256, 0, stream>>>(Xf32, xh, nx);
    k_cvt_wt<<<cdiv(128 * 128, 256), 256, 0, stream>>>(W, wth);
    k_gemm_att<<<cdiv(N, 16), 256, 0, stream>>>(xh, wth, attS, attD, Hlin, a_src, a_dst, N);
    k_init<<<cdiv(nx, 256), 256, 0, stream>>>(segmax, segsum, accum, N * 8, nx);
    const long long tot = (long long)(E + N) * 8;
    const int eb = cdiv(tot, 256);
    k_edge_max <<<eb, 256, 0, stream>>>(srcI, dstI, E, N, a_src, a_dst, segmax);
    k_edge_sum <<<eb, 256, 0, stream>>>(srcI, dstI, E, N, a_src, a_dst, segmax, segsum);
    k_edge_aggr<<<eb, 256, 0, stream>>>(srcI, dstI, E, N, a_src, a_dst, segmax, segsum, Hlin, accum);
}

extern "C" void kernel_launch(void* const* d_in, const int* in_sizes, int n_in,
                              void* d_out, int out_size, void* d_ws, size_t ws_size,
                              hipStream_t stream) {
    const float* x    = (const float*)d_in[0];
    const int*   eidx = (const int*)  d_in[1];
    const float* W0   = (const float*)d_in[2];
    const float* as0  = (const float*)d_in[3];
    const float* ad0  = (const float*)d_in[4];
    // d_in[5] = b0 : provably cancelled by batch-norm (per-column shift) -> skipped
    const float* g0   = (const float*)d_in[6];
    const float* be0  = (const float*)d_in[7];
    const float* W1   = (const float*)d_in[8];
    const float* as1  = (const float*)d_in[9];
    const float* ad1  = (const float*)d_in[10];
    // d_in[11] = b1 : cancelled by batch-norm -> skipped
    const float* g1   = (const float*)d_in[12];
    const float* be1  = (const float*)d_in[13];
    const float* Wc   = (const float*)d_in[14];
    const float* bc   = (const float*)d_in[15];
    float* out = (float*)d_out;
    (void)n_in; (void)out_size; (void)ws_size;

    const int N = in_sizes[0] / 128;
    const int E = in_sizes[1] / 2;
    const int* srcI = eidx;        // edge_index row 0
    const int* dstI = eidx + E;    // edge_index row 1

    // --- carve workspace (256B aligned) ---
    char* w = (char*)d_ws;
    auto carve = [&](size_t bytes) -> void* {
        void* p = (void*)w;
        w += (bytes + 255) & ~(size_t)255;
        return p;
    };
    _Float16* xh     = (_Float16*)carve((size_t)N * 128 * sizeof(_Float16));
    _Float16* wth    = (_Float16*)carve(128 * 128 * sizeof(_Float16));
    float*    Hlin   = (float*)carve((size_t)N * 128 * sizeof(float));
    float*    a_src  = (float*)carve((size_t)N * 8 * sizeof(float));
    float*    a_dst  = (float*)carve((size_t)N * 8 * sizeof(float));
    int*      segmax = (int*)  carve((size_t)N * 8 * sizeof(int));
    float*    segsum = (float*)carve((size_t)N * 8 * sizeof(float));
    float*    accum  = (float*)carve((size_t)N * 128 * sizeof(float));
    float*    h0     = (float*)carve((size_t)N * 128 * sizeof(float));
    float*    hm     = (float*)carve((size_t)N * 16 * sizeof(float));
    float*    colsum = (float*)carve(128 * sizeof(float));
    float*    colsq  = (float*)carve(128 * sizeof(float));

    // ===== layer 0 =====
    run_layer(x, W0, as0, ad0, xh, wth, Hlin, a_src, a_dst, segmax, segsum, accum,
              srcI, dstI, N, E, stream);
    k_zero2<<<1, 256, 0, stream>>>(colsum, colsq, 128);
    k_colstats<128><<<512, 256, 0, stream>>>(accum, N, colsum, colsq);
    k_bn_elu<<<cdiv((long long)N * 128, 256), 256, 0, stream>>>(accum, colsum, colsq, g0, be0, h0, N);

    // ===== layer 1 =====
    run_layer(h0, W1, as1, ad1, xh, wth, Hlin, a_src, a_dst, segmax, segsum, accum,
              srcI, dstI, N, E, stream);
    k_head_mean<<<cdiv(N * 16, 256), 256, 0, stream>>>(accum, hm, N);
    k_zero2<<<1, 256, 0, stream>>>(colsum, colsq, 16);
    k_colstats<16><<<512, 256, 0, stream>>>(hm, N, colsum, colsq);
    k_final<<<cdiv(N, 256), 256, 0, stream>>>(hm, colsum, colsq, g1, be1, Wc, bc, out, N);
}
